// FlashDeepseekV2Attention_15418932592645
// MI455X (gfx1250) — matrix operand order repaired and checked by reading, not compiled, not run on hardware
//
#include <hip/hip_runtime.h>
#include <math.h>

// ---------------------------------------------------------------------------
// DeepSeek-V2 MLA forward for MI455X (gfx1250, wave32, WMMA 16x16x32 f16).
// GEMM B-tiles staged into LDS by the Tensor Data Mover (TDM), triple-buffered
// so the DMA of tile n+1 overlaps the WMMAs of tile n (TENSORcnt pipelining).
// ---------------------------------------------------------------------------

#define USE_TDM 1          // 0 -> fall back to per-thread async global->LDS

typedef __attribute__((ext_vector_type(16))) _Float16 v16h;
typedef __attribute__((ext_vector_type(8)))  _Float16 v8h;
typedef __attribute__((ext_vector_type(8)))  float    v8f;
typedef __attribute__((ext_vector_type(4)))  unsigned int v4u;
typedef __attribute__((ext_vector_type(8)))  int v8i;
typedef __attribute__((ext_vector_type(4)))  int v4i;

#define S_LEN 2048
#define HID   2048
#define NH    16
#define QL    1536
#define KVL   512
#define DR    64
#define DN    128
#define DV    128
#define QHD   192   // DN + DR

// ===========================================================================
// TDM: DMA a 2D tile (64 rows x 32 f16, row stride = K elements) from global
// memory into contiguous LDS. One descriptor, wave-level, TENSORcnt-tracked.
// ===========================================================================
#if USE_TDM
__device__ __forceinline__ void tdm_load_b_tile(const _Float16* gsrc,
                                                unsigned lds_byte, int K)
{
    unsigned long long ga = (unsigned long long)(uintptr_t)gsrc;
    v4u g0;
    g0[0] = 1u;                                         // count=1 (valid D#)
    g0[1] = lds_byte;                                   // lds_addr
    g0[2] = (unsigned)(ga & 0xffffffffu);               // global_addr[31:0]
    g0[3] = (unsigned)((ga >> 32) & 0x1ffffffu)         // global_addr[56:32]
          | (2u << 30);                                 // type=2 ("image")
    v8i g1;
    g1[0] = 0x00010000;                 // data_size=1 (2 bytes/elem)
    g1[1] = (int)(((unsigned)K & 0xffffu) << 16);   // tensor_dim0[15:0]
    g1[2] = (int)(64u << 16);           // tensor_dim0[31:16]=0 | tensor_dim1=64
    g1[3] = (int)(32u << 16);           // tensor_dim1[31:16]=0 | tile_dim0=32
    g1[4] = 64;                         // tile_dim1=64 | tile_dim2=0
    g1[5] = K;                          // tensor_dim0_stride[31:0]
    g1[6] = 0;                          // stride[47:32]=0 | dim1_stride[15:0]=0
    g1[7] = 0;
    v4i g2 = {0, 0, 0, 0};
    v4i g3 = {0, 0, 0, 0};
#if __clang_major__ >= 23
    v8i g4 = {0, 0, 0, 0, 0, 0, 0, 0};
    __builtin_amdgcn_tensor_load_to_lds(g0, g1, g2, g3, g4, 0);
#else
    __builtin_amdgcn_tensor_load_to_lds(g0, g1, g2, g3, 0);
#endif
}
#endif

// ===========================================================================
// Elementwise f32 -> f16 (n must be a multiple of 1024).
// ===========================================================================
__global__ __launch_bounds__(256)
void cvt_f16(const float* __restrict__ in, _Float16* __restrict__ out)
{
    int i = (blockIdx.x * 256 + threadIdx.x) * 4;
    float4 v = *(const float4*)(in + i);
    out[i]     = (_Float16)v.x;
    out[i + 1] = (_Float16)v.y;
    out[i + 2] = (_Float16)v.z;
    out[i + 3] = (_Float16)v.w;
}

// ===========================================================================
// Transpose + convert: in f32 (R x C) -> out f16 (C x R). R,C multiples of 32.
// ===========================================================================
__global__ __launch_bounds__(256)
void transpose_f16(const float* __restrict__ in, _Float16* __restrict__ outT,
                   int R, int C)
{
    __shared__ float tile[32][33];
    const int c0 = blockIdx.x * 32, r0 = blockIdx.y * 32;
    const int tx = threadIdx.x & 31, ty = threadIdx.x >> 5;   // 32 x 8
#pragma unroll
    for (int i = 0; i < 32; i += 8)
        tile[ty + i][tx] = in[(size_t)(r0 + ty + i) * C + c0 + tx];
    __syncthreads();
#pragma unroll
    for (int i = 0; i < 32; i += 8)
        outT[(size_t)(c0 + ty + i) * R + r0 + tx] = (_Float16)tile[tx][ty + i];
}

// ===========================================================================
// WMMA GEMM: C[M,N](f32) = A[M,K](f16, row-major) * Bt[N,K](f16, row-major).
// block (32,8): 8 waves, each owning TWO 16-row tiles -> 256 rows per block;
// 64 cols per block. B^T tile (64x32) staged in triple-buffered LDS by the
// TDM: tile n+1's DMA overlaps tile n's WMMAs; one barrier per K-step.
// grid (N/64, M/256).
// ===========================================================================
__global__ __launch_bounds__(256)
void gemm_f16(const _Float16* __restrict__ A, const _Float16* __restrict__ Bt,
              float* __restrict__ C, int M, int N, int K)
{
    const int tid    = threadIdx.y * 32 + threadIdx.x;
    const int lane   = threadIdx.x;
    const int half16 = lane >> 4;
    const int lr     = lane & 15;
    const int row0   = blockIdx.y * 256 + threadIdx.y * 32 + lr;  // tile0 row
    const int bn0    = blockIdx.x * 64;

    __shared__ _Float16 bs[3][64][32];                 // triple-buffered B^T

    const _Float16* arow0 = A + (size_t)row0 * K + (half16 << 3);
    const _Float16* arow1 = arow0 + (size_t)16 * K;

    const _Float16* btile = Bt + (size_t)bn0 * K;      // tile base (row bn0)
    const unsigned  lds0  = (unsigned)(uintptr_t)&bs[0][0][0];
    const bool dmaWave = (threadIdx.y == 0);
#if !USE_TDM
    const int ln = tid >> 2;
    const int lk = (tid & 3) << 3;
    const _Float16* bsrc = Bt + (size_t)(bn0 + ln) * K + lk;
    const unsigned ldsthr = (unsigned)(uintptr_t)&bs[0][ln][lk];
    (void)dmaWave; (void)lds0; (void)btile;
#endif

    v8f z = {0.f,0.f,0.f,0.f,0.f,0.f,0.f,0.f};
    v8f acc0[4] = {z, z, z, z};
    v8f acc1[4] = {z, z, z, z};

    const int T = K >> 5;                 // number of 32-wide K tiles
    int cur = 0, nxt = 1;

#if USE_TDM
    if (dmaWave) tdm_load_b_tile(btile, lds0, K);
#else
    {
        unsigned long long ga = (unsigned long long)(uintptr_t)bsrc;
        asm volatile("global_load_async_to_lds_b128 %0, %1, off"
                     :: "v"(ldsthr), "v"(ga) : "memory");
    }
#endif

    for (int n = 0; n < T; ++n) {
        const int k0 = n << 5;
#if USE_TDM
        if (dmaWave) {
            if (n + 1 < T) {
                tdm_load_b_tile(btile + k0 + 32, lds0 + (unsigned)nxt * 4096u, K);
                __builtin_amdgcn_s_wait_tensorcnt(1);
            } else {
                __builtin_amdgcn_s_wait_tensorcnt(0);
            }
        }
#else
        if (n + 1 < T) {
            unsigned long long ga = (unsigned long long)(uintptr_t)(bsrc + k0 + 32);
            unsigned dst = ldsthr + (unsigned)nxt * 4096u;
            asm volatile("global_load_async_to_lds_b128 %0, %1, off"
                         :: "v"(dst), "v"(ga) : "memory");
            asm volatile("s_wait_asynccnt 0x1" ::: "memory");
        } else {
            asm volatile("s_wait_asynccnt 0x0" ::: "memory");
        }
#endif
        __syncthreads();
        __builtin_prefetch(arow0 + k0 + 32, 0, 1);

        v8h a00 = *(const v8h*)(arow0 + k0);
        v8h a01 = *(const v8h*)(arow0 + k0 + 16);
        v8h a10 = *(const v8h*)(arow1 + k0);
        v8h a11 = *(const v8h*)(arow1 + k0 + 16);
        v16h a0, a1;
#pragma unroll
        for (int h = 0; h < 8; ++h) {
            a0[h] = a00[h]; a0[8 + h] = a01[h];
            a1[h] = a10[h]; a1[8 + h] = a11[h];
        }

#pragma unroll
        for (int j = 0; j < 4; ++j) {
            const _Float16* bp = &bs[cur][j * 16 + lr][half16 << 3];
            v8h b0 = *(const v8h*)bp;
            v8h b1 = *(const v8h*)(bp + 16);
            v16h b;
#pragma unroll
            for (int h = 0; h < 8; ++h) { b[h] = b0[h]; b[8 + h] = b1[h]; }
            acc0[j] = __builtin_amdgcn_wmma_f32_16x16x32_f16(
                false, a0, false, b, (short)0, acc0[j], false, false);
            acc1[j] = __builtin_amdgcn_wmma_f32_16x16x32_f16(
                false, a1, false, b, (short)0, acc1[j], false, false);
        }
        cur = (cur == 2) ? 0 : cur + 1;
        nxt = (nxt == 2) ? 0 : nxt + 1;
        // triple buffering bounds barrier skew: no trailing barrier needed
    }

    const int crow = blockIdx.y * 256 + threadIdx.y * 32 + (half16 << 3);
#pragma unroll
    for (int j = 0; j < 4; ++j) {
        float* cp0 = C + (size_t)crow * N + bn0 + j * 16 + lr;
        float* cp1 = cp0 + (size_t)16 * N;
#pragma unroll
        for (int r = 0; r < 8; ++r) {
            cp0[(size_t)r * N] = acc0[j][r];
            cp1[(size_t)r * N] = acc1[j][r];
        }
    }
}

// ===========================================================================
// RMSNorm: f32 row in -> f16 row out. One block per row.
// ===========================================================================
__global__ __launch_bounds__(256)
void rmsnorm_f16(const float* __restrict__ x, const float* __restrict__ w,
                 _Float16* __restrict__ y, int n)
{
    const float* row = x + (size_t)blockIdx.x * n;
    float ss = 0.f;
    for (int i = threadIdx.x; i < n; i += 256) { float v = row[i]; ss += v * v; }
    __shared__ float red[256];
    red[threadIdx.x] = ss;
    __syncthreads();
    for (int s = 128; s > 0; s >>= 1) {
        if (threadIdx.x < s) red[threadIdx.x] += red[threadIdx.x + s];
        __syncthreads();
    }
    float inv = rsqrtf(red[0] / (float)n + 1e-6f);
    _Float16* out = y + (size_t)blockIdx.x * n;
    for (int i = threadIdx.x; i < n; i += 256) out[i] = (_Float16)(row[i] * inv * w[i]);
}

// ===========================================================================
// ckv (S,576) f32 -> rmsnorm'd kv latent (f16, S x 512) + rope'd k_pe (f16, S x 64).
// ===========================================================================
__global__ __launch_bounds__(256)
void kv_norm_rope_f16(const float* __restrict__ ckv, const float* __restrict__ w,
                      _Float16* __restrict__ kv_ln, _Float16* __restrict__ k_pe)
{
    const int pos = blockIdx.x;
    const float* row = ckv + (size_t)pos * (KVL + DR);
    float ss = 0.f;
    for (int i = threadIdx.x; i < KVL; i += 256) { float v = row[i]; ss += v * v; }
    __shared__ float red[256];
    red[threadIdx.x] = ss;
    __syncthreads();
    for (int s = 128; s > 0; s >>= 1) {
        if (threadIdx.x < s) red[threadIdx.x] += red[threadIdx.x + s];
        __syncthreads();
    }
    float inv = rsqrtf(red[0] / (float)KVL + 1e-6f);
    _Float16* out = kv_ln + (size_t)pos * KVL;
    for (int i = threadIdx.x; i < KVL; i += 256) out[i] = (_Float16)(row[i] * inv * w[i]);

    if (threadIdx.x < 32) {
        int j = threadIdx.x;
        float f = (float)pos * powf(10000.f, -(float)j * (1.f / 32.f));
        float c = cosf(f), s = sinf(f);
        float x1 = row[KVL + j];
        float x2 = row[KVL + 32 + j];
        k_pe[(size_t)pos * DR + j]      = (_Float16)(x1 * c - x2 * s);
        k_pe[(size_t)pos * DR + 32 + j] = (_Float16)(x2 * c + x1 * s);
    }
}

// ===========================================================================
// q (S,NH,192) f32 -> rope(q_pe) + softmax-scale folded -> f16. grid (S,NH).
// ===========================================================================
__global__ __launch_bounds__(192)
void q_rope_scale_f16(const float* __restrict__ qbuf, _Float16* __restrict__ qh)
{
    const int pos = blockIdx.x, head = blockIdx.y, d = threadIdx.x;
    const float scale = 0.07216878364870323f;          // 192^-0.5
    const float* src = qbuf + ((size_t)pos * NH + head) * QHD;
    _Float16*    dst = qh   + ((size_t)pos * NH + head) * QHD;
    if (d < DN) {
        dst[d] = (_Float16)(src[d] * scale);
    } else if (d < DN + 32) {
        int j = d - DN;
        float f = (float)pos * powf(10000.f, -(float)j * (1.f / 32.f));
        float c = cosf(f), s = sinf(f);
        float x1 = src[DN + j], x2 = src[DN + 32 + j];
        dst[DN + j]      = (_Float16)((x1 * c - x2 * s) * scale);
        dst[DN + 32 + j] = (_Float16)((x2 * c + x1 * s) * scale);
    }
}

// ===========================================================================
// kvbuf (S,NH,256) f32 -> knh (S,NH,128) f16 (the k_nope part).
// ===========================================================================
__global__ __launch_bounds__(256)
void kv_extract_f16(const float* __restrict__ kvbuf, _Float16* __restrict__ knh)
{
    size_t i = (size_t)blockIdx.x * 256 + threadIdx.x;   // over S*NH*128
    size_t row = i >> 7, d = i & 127;
    knh[i] = (_Float16)kvbuf[row * 256 + d];
}

// ===========================================================================
// kvbuf (S,NH,256) f32, v = [...,128:256] -> vt (NH*128, S) f16, transposed.
// grid (DV/32, S/32, NH), block 256.
// ===========================================================================
__global__ __launch_bounds__(256)
void v_transpose_f16(const float* __restrict__ kvbuf, _Float16* __restrict__ vt)
{
    __shared__ float tile[32][33];
    const int head = blockIdx.z;
    const int d0 = blockIdx.x * 32, s0 = blockIdx.y * 32;
    const int tx = threadIdx.x & 31, ty = threadIdx.x >> 5;
#pragma unroll
    for (int i = 0; i < 32; i += 8)
        tile[ty + i][tx] = kvbuf[((size_t)(s0 + ty + i) * NH + head) * (DN + DV) + DN + d0 + tx];
    __syncthreads();
#pragma unroll
    for (int i = 0; i < 32; i += 8)
        vt[(size_t)(head * DV + d0 + ty + i) * S_LEN + s0 + tx] = (_Float16)tile[tx][ty + i];
}

// ===========================================================================
// Flash-style causal MLA attention. One wave per (16-query tile, head).
//   qh  : (S, NH, 192) f16, pre-scaled + rope'd
//   knh : (S, NH, 128) f16      kpeh : (S, 64) f16 (shared across heads)
//   vt  : (NH*128, S)  f16 (transposed V)
//   out : (S, NH*128)  f16 (feeds the f16 output-projection GEMM directly)
// ===========================================================================
__global__ __launch_bounds__(32)
void mla_attention(const _Float16* __restrict__ qh, const _Float16* __restrict__ knh,
                   const _Float16* __restrict__ kpeh, const _Float16* __restrict__ vt,
                   _Float16* __restrict__ out)
{
    const int lane   = threadIdx.x;
    const int half16 = lane >> 4;
    const int lr     = lane & 15;
    const int rowb   = half16 << 3;
    const int q0     = blockIdx.x * 16;
    const int head   = blockIdx.y;

    // Q fragments: 6 x (16x32) covering the 192-dim head.
    v16h qa[6];
    {
        const _Float16* qrow = qh + ((size_t)(q0 + lr) * NH + head) * QHD + (half16 << 4);
#pragma unroll
        for (int t = 0; t < 6; ++t) {
            v8h a0 = *(const v8h*)(qrow + t * 32);
            v8h a1 = *(const v8h*)(qrow + t * 32 + 8);
#pragma unroll
            for (int h = 0; h < 8; ++h) { qa[t][h] = a0[h]; qa[t][8 + h] = a1[h]; }
        }
    }

    v8f z = {0.f,0.f,0.f,0.f,0.f,0.f,0.f,0.f};
    v8f o[8] = {z, z, z, z, z, z, z, z};
    float m_r[8], l_r[8];
#pragma unroll
    for (int r = 0; r < 8; ++r) { m_r[r] = -1e30f; l_r[r] = 0.f; }

    __shared__ _Float16 pbuf[16 * 32];

    for (int kb = 0; kb < q0 + 16; kb += 32) {
        // ---- S = Q K^T over two 16-key tiles (keys never exceed S) -------
        v8f s0 = z, s1 = z;
        const int key_a = kb + lr;
        const int key_b = kb + 16 + lr;
        const _Float16* ka = knh + ((size_t)key_a * NH + head) * DN + (half16 << 4);
        const _Float16* kc = knh + ((size_t)key_b * NH + head) * DN + (half16 << 4);
#pragma unroll
        for (int t = 0; t < 4; ++t) {          // k_nope dims 0..127
            v16h b;
            v8h b0 = *(const v8h*)(ka + t * 32);
            v8h b1 = *(const v8h*)(ka + t * 32 + 8);
#pragma unroll
            for (int h = 0; h < 8; ++h) { b[h] = b0[h]; b[8 + h] = b1[h]; }
            s0 = __builtin_amdgcn_wmma_f32_16x16x32_f16(
                false, qa[t], false, b, (short)0, s0, false, false);
            v8h c0 = *(const v8h*)(kc + t * 32);
            v8h c1 = *(const v8h*)(kc + t * 32 + 8);
#pragma unroll
            for (int h = 0; h < 8; ++h) { b[h] = c0[h]; b[8 + h] = c1[h]; }
            s1 = __builtin_amdgcn_wmma_f32_16x16x32_f16(
                false, qa[t], false, b, (short)0, s1, false, false);
        }
        const _Float16* pa_ = kpeh + (size_t)key_a * DR + (half16 << 4);
        const _Float16* pc_ = kpeh + (size_t)key_b * DR + (half16 << 4);
#pragma unroll
        for (int t = 4; t < 6; ++t) {          // k_pe dims 128..191
            const int off = (t - 4) * 32;
            v16h b;
            v8h b0 = *(const v8h*)(pa_ + off);
            v8h b1 = *(const v8h*)(pa_ + off + 8);
#pragma unroll
            for (int h = 0; h < 8; ++h) { b[h] = b0[h]; b[8 + h] = b1[h]; }
            s0 = __builtin_amdgcn_wmma_f32_16x16x32_f16(
                false, qa[t], false, b, (short)0, s0, false, false);
            v8h c0 = *(const v8h*)(pc_ + off);
            v8h c1 = *(const v8h*)(pc_ + off + 8);
#pragma unroll
            for (int h = 0; h < 8; ++h) { b[h] = c0[h]; b[8 + h] = c1[h]; }
            s1 = __builtin_amdgcn_wmma_f32_16x16x32_f16(
                false, qa[t], false, b, (short)0, s1, false, false);
        }

        // ---- causal mask + online softmax --------------------------------
#pragma unroll
        for (int r = 0; r < 8; ++r) {
            const int qi = q0 + rowb + r;
            s0[r] = (kb + lr      > qi) ? -1e30f : s0[r];
            s1[r] = (kb + 16 + lr > qi) ? -1e30f : s1[r];
        }
#pragma unroll
        for (int r = 0; r < 8; ++r) {
            float v = fmaxf(s0[r], s1[r]);
#pragma unroll
            for (int off = 8; off > 0; off >>= 1) v = fmaxf(v, __shfl_xor(v, off, 16));
            float mn    = fmaxf(m_r[r], v);
            float alpha = __expf(m_r[r] - mn);
            float p0 = __expf(s0[r] - mn);
            float p1 = __expf(s1[r] - mn);
            float rs = p0 + p1;
#pragma unroll
            for (int off = 8; off > 0; off >>= 1) rs += __shfl_xor(rs, off, 16);
            l_r[r] = l_r[r] * alpha + rs;
            m_r[r] = mn;
            s0[r] = p0; s1[r] = p1;
#pragma unroll
            for (int j = 0; j < 8; ++j) o[j][r] *= alpha;
        }

        // ---- re-stripe P (C layout) -> A layout via LDS ------------------
        __syncthreads();
#pragma unroll
        for (int r = 0; r < 8; ++r) {
            pbuf[(rowb + r) * 32 + lr]      = (_Float16)s0[r];
            pbuf[(rowb + r) * 32 + 16 + lr] = (_Float16)s1[r];
        }
        __syncthreads();
        v16h pfrag;
#pragma unroll
        for (int h = 0; h < 8; ++h) pfrag[h]     = pbuf[lr * 32 + (half16 << 4) + h];
#pragma unroll
        for (int h = 0; h < 8; ++h) pfrag[8 + h] = pbuf[lr * 32 + (half16 << 4) + 8 + h];

        // ---- O += P V (V transposed: contiguous along keys) --------------
#pragma unroll
        for (int j = 0; j < 8; ++j) {
            const _Float16* vp = vt + (size_t)(head * DV + j * 16 + lr) * S_LEN
                                    + kb + (half16 << 4);
            v8h b0 = *(const v8h*)vp;
            v8h b1 = *(const v8h*)(vp + 8);
            v16h b;
#pragma unroll
            for (int h = 0; h < 8; ++h) { b[h] = b0[h]; b[8 + h] = b1[h]; }
            o[j] = __builtin_amdgcn_wmma_f32_16x16x32_f16(
                false, pfrag, false, b, (short)0, o[j], false, false);
        }
    }

    // ---- normalize + write (S, NH*DV) f16 ---------------------------------
#pragma unroll
    for (int r = 0; r < 8; ++r) {
        float inv = 1.f / l_r[r];
        size_t rowoff = (size_t)(q0 + rowb + r) * (NH * DV) + head * DV + lr;
#pragma unroll
        for (int j = 0; j < 8; ++j) out[rowoff + j * 16] = (_Float16)(o[j][r] * inv);
    }
}

// ===========================================================================
// Host-side launcher.
// ===========================================================================
extern "C" void kernel_launch(void* const* d_in, const int* in_sizes, int n_in,
                              void* d_out, int out_size, void* d_ws, size_t ws_size,
                              hipStream_t stream)
{
    (void)in_sizes; (void)n_in; (void)out_size; (void)ws_size;

    const float* x       = (const float*)d_in[0];
    const float* wq_a    = (const float*)d_in[1];
    const float* q_ln_w  = (const float*)d_in[2];
    const float* wq_b    = (const float*)d_in[3];
    const float* wkv_a   = (const float*)d_in[4];
    const float* kv_ln_w = (const float*)d_in[5];
    const float* wkv_b   = (const float*)d_in[6];
    const float* wo      = (const float*)d_in[7];
    float* out = (float*)d_out;

    // ---- workspace carve-up (f32 region first, then f16) ------------------
    float* fp = (float*)d_ws;
    float* q_lat = fp;  fp += (size_t)S_LEN * QL;              // S x 1536
    float* ckv   = fp;  fp += (size_t)S_LEN * (KVL + DR);      // S x 576
    float* qbuf  = fp;  fp += (size_t)S_LEN * NH * QHD;        // S x 3072
    float* kvbuf = fp;  fp += (size_t)S_LEN * NH * (DN + DV);  // S x 4096

    _Float16* hp = (_Float16*)fp;
    _Float16* xh     = hp;  hp += (size_t)S_LEN * HID;
    _Float16* wqaT   = hp;  hp += (size_t)QL * HID;
    _Float16* wkvaT  = hp;  hp += (size_t)(KVL + DR) * HID;
    _Float16* wqbT   = hp;  hp += (size_t)(NH * QHD) * QL;
    _Float16* wkvbT  = hp;  hp += (size_t)(NH * (DN + DV)) * KVL;
    _Float16* woT    = hp;  hp += (size_t)HID * HID;
    _Float16* qlath  = hp;  hp += (size_t)S_LEN * QL;
    _Float16* kvlnh  = hp;  hp += (size_t)S_LEN * KVL;
    _Float16* kpeh   = hp;  hp += (size_t)S_LEN * DR;
    _Float16* qh     = hp;  hp += (size_t)S_LEN * NH * QHD;
    _Float16* knh    = hp;  hp += (size_t)S_LEN * NH * DN;
    _Float16* vth    = hp;  hp += (size_t)NH * DV * S_LEN;
    _Float16* atth   = hp;  hp += (size_t)S_LEN * HID;

    dim3 gblk(32, 8, 1);

    // ---- 0) precision prep: f16 activations, f16 transposed weights -------
    cvt_f16<<<(S_LEN * HID) / 1024, 256, 0, stream>>>(x, xh);
    transpose_f16<<<dim3(QL / 32, HID / 32), 256, 0, stream>>>(wq_a, wqaT, HID, QL);
    transpose_f16<<<dim3((KVL + DR) / 32, HID / 32), 256, 0, stream>>>(wkv_a, wkvaT, HID, KVL + DR);
    transpose_f16<<<dim3((NH * QHD) / 32, QL / 32), 256, 0, stream>>>(wq_b, wqbT, QL, NH * QHD);
    transpose_f16<<<dim3((NH * (DN + DV)) / 32, KVL / 32), 256, 0, stream>>>(wkv_b, wkvbT, KVL, NH * (DN + DV));
    transpose_f16<<<dim3(HID / 32, HID / 32), 256, 0, stream>>>(wo, woT, HID, HID);

    // ---- 1) down-projections ----------------------------------------------
    gemm_f16<<<dim3(QL / 64, S_LEN / 256), gblk, 0, stream>>>(xh, wqaT, q_lat, S_LEN, QL, HID);
    gemm_f16<<<dim3((KVL + DR) / 64, S_LEN / 256), gblk, 0, stream>>>(xh, wkvaT, ckv, S_LEN, KVL + DR, HID);

    // ---- 2) norms + k_pe rope (f16 out) -----------------------------------
    rmsnorm_f16<<<S_LEN, 256, 0, stream>>>(q_lat, q_ln_w, qlath, QL);
    kv_norm_rope_f16<<<S_LEN, 256, 0, stream>>>(ckv, kv_ln_w, kvlnh, kpeh);

    // ---- 3) up-projections --------------------------------------------------
    gemm_f16<<<dim3((NH * QHD) / 64, S_LEN / 256), gblk, 0, stream>>>(qlath, wqbT, qbuf, S_LEN, NH * QHD, QL);
    gemm_f16<<<dim3((NH * (DN + DV)) / 64, S_LEN / 256), gblk, 0, stream>>>(kvlnh, wkvbT, kvbuf, S_LEN, NH * (DN + DV), KVL);

    // ---- 4) attention operand prep ----------------------------------------
    q_rope_scale_f16<<<dim3(S_LEN, NH), 192, 0, stream>>>(qbuf, qh);
    kv_extract_f16<<<(S_LEN * NH * DN) / 256, 256, 0, stream>>>(kvbuf, knh);
    v_transpose_f16<<<dim3(DV / 32, S_LEN / 32, NH), 256, 0, stream>>>(kvbuf, vth);

    // ---- 5) causal attention (writes f16 directly) ------------------------
    mla_attention<<<dim3(S_LEN / 16, NH), 32, 0, stream>>>(qh, knh, kpeh, vth, atth);

    // ---- 6) output projection -> d_out ------------------------------------
    gemm_f16<<<dim3(HID / 64, S_LEN / 256), gblk, 0, stream>>>(atth, woT, out, S_LEN, HID, HID);
}
